// SAModule_67250597921401
// MI455X (gfx1250) — matrix operand order repaired and checked by reading, not compiled, not run on hardware
//
#include <hip/hip_runtime.h>
#include <hip/hip_bf16.h>

typedef __attribute__((ext_vector_type(16))) _Float16 v16h;
typedef __attribute__((ext_vector_type(8)))  float    v8f;

#define N_PTS   16384
#define D_FEAT  64
#define M_CTR   4096
#define K_NBR   32
#define H_DIM   128
#define OUT_DIM 128
#define EIN     132           // 2*D + 4
#define KPAD    160           // EIN padded to 5 * 32
#define R2      0.0625f
#define NEGV    (-1e30f)

// ---- workspace layout (bytes) ----
#define OFF_IDX  0u                        // int   [4096]
#define OFF_COL  16384u                    // int   [4096*32]
#define OFF_BD   540672u                   // float [4096*32]
#define OFF_W1F  1064960u                  // f16   [20480]  W1 in WMMA-B fragment order (K padded)
#define OFF_W2F  1105920u                  // f16   [16384]  W2 in WMMA-B fragment order
#define OFF_AGG  1138688u                  // f16   [4096*128] agg in WMMA-A fragment order (per 128-row block)

// async Global->LDS staging: each lane moves 16B; lds addr in %0, 32-bit voffset in %1, 64-bit sgpr base in %2
#define ASYNC_LD_B128(ldsaddr, voff, gbase)                                   \
    asm volatile("global_load_async_to_lds_b128 %0, %1, %2 offset:0"          \
                 :: "v"(ldsaddr), "v"(voff), "s"(gbase) : "memory")
#define ASYNC_WAIT() asm volatile("s_wait_asynccnt 0x0" ::: "memory")

// ---------------------------------------------------------------------------
// fragment-order addressing helpers (v_wmma_f32_16x16x32_f16)
//  A (16x32): lane = (M%16) + 16*hi ; lane's 16 halves contiguous.
//    element (row,k): ko=k%32; hi=(ko>>3)&1; idx=(ko&7)+((ko>>4)&1)*8
//  B (32x16): lane = (N%16) + 16*hi ; hi=(ko>>3)&1 ; idx=(ko&7)+((ko>>4)&1)*8 (same as A)
// ---------------------------------------------------------------------------
__device__ __forceinline__ int eaA_addr(int row, int k, int nkc) {
    int m  = row >> 4, lr = row & 15;
    int kc = k >> 5,  ko = k & 31;
    int hi  = (ko >> 3) & 1;
    int idx = (ko & 7) + ((ko >> 4) & 1) * 8;
    return (((m * nkc + kc) * 32) + (lr + 16 * hi)) * 16 + idx;
}

__device__ __forceinline__ float safe_norm3(float ax, float ay, float az) {
    float s = ax*ax + ay*ay + az*az;
    return s > 0.f ? sqrtf(s) : 0.f;
}
__device__ __forceinline__ float get_angle3(float ax, float ay, float az,
                                            float bx, float by, float bz) {
    float cxv = ay*bz - az*by;
    float cyv = az*bx - ax*bz;
    float czv = ax*by - ay*bx;
    float cn  = safe_norm3(cxv, cyv, czv);
    float dot = ax*bx + ay*by + az*bz;
    if (cn == 0.f && dot == 0.f) dot = 1.f;
    return atan2f(cn, dot);
}

// ---------------------------------------------------------------------------
// kernel 0: W1 -> fragment-order f16 (K padded to 160), W2 -> fragment-order f16
// ---------------------------------------------------------------------------
__global__ void k_convert(const float* __restrict__ W1, const float* __restrict__ W2,
                          _Float16* __restrict__ W1f, _Float16* __restrict__ W2f) {
    int tid = blockIdx.x * blockDim.x + threadIdx.x;
    if (tid < H_DIM * KPAD) {                       // 20480 = 8nt * 5kc * 32lane * 16idx
        int idx  = tid & 15;
        int lane = (tid >> 4) & 31;
        int rest = tid >> 9;
        int kc = rest % 5, nt = rest / 5;
        int n  = nt * 16 + (lane & 15);
        int k  = kc * 32 + ((lane >= 16) ? 8 : 0) + (idx & 7) + ((idx >> 3) << 4);
        W1f[tid] = (k < EIN) ? (_Float16)W1[k * H_DIM + n] : (_Float16)0.f;
    } else if (tid < H_DIM * KPAD + OUT_DIM * H_DIM) {
        int t2 = tid - H_DIM * KPAD;                // 16384 = 8nt * 4kc * 32lane * 16idx
        int idx  = t2 & 15;
        int lane = (t2 >> 4) & 31;
        int rest = t2 >> 9;
        int kc = rest & 3, nt = rest >> 2;
        int n  = nt * 16 + (lane & 15);
        int k  = kc * 32 + ((lane >= 16) ? 8 : 0) + (idx & 7) + ((idx >> 3) << 4);
        W2f[t2] = (_Float16)W2[k * OUT_DIM + n];
    }
}

// ---------------------------------------------------------------------------
// kernel 1: farthest point sampling — single block, 1024 threads (32 waves)
// ---------------------------------------------------------------------------
__global__ void k_fps(const float* __restrict__ pos, int* __restrict__ wsIdx) {
    __shared__ float rv[32];
    __shared__ int   ri[32];
    __shared__ int   shLast;
    const int t = threadIdx.x, wv = t >> 5, lane = t & 31;

    float dist[16];
#pragma unroll
    for (int s = 0; s < 16; ++s) dist[s] = 3.0e38f;
    if (t == 0) shLast = 0;
    __syncthreads();

    for (int m = 0; m < M_CTR; ++m) {
        int last = shLast;
        if (t == 0) wsIdx[m] = last;
        float lx = pos[last*3+0], ly = pos[last*3+1], lz = pos[last*3+2];

        float bv = -1.f; int bi = 0x7fffffff;
#pragma unroll
        for (int s = 0; s < 16; ++s) {
            int i = t + (s << 10);
            float dx = pos[i*3+0]-lx, dy = pos[i*3+1]-ly, dz = pos[i*3+2]-lz;
            float d  = dx*dx + dy*dy + dz*dz;
            float nd = dist[s] < d ? dist[s] : d;
            dist[s] = nd;
            if (nd > bv || (nd == bv && i < bi)) { bv = nd; bi = i; }
        }
#pragma unroll
        for (int off = 16; off; off >>= 1) {
            float ov = __shfl_xor(bv, off);
            int   oi = __shfl_xor(bi, off);
            if (ov > bv || (ov == bv && oi < bi)) { bv = ov; bi = oi; }
        }
        if (lane == 0) { rv[wv] = bv; ri[wv] = bi; }
        __syncthreads();
        if (wv == 0) {
            float v = rv[lane]; int i2 = ri[lane];
#pragma unroll
            for (int off = 16; off; off >>= 1) {
                float ov = __shfl_xor(v, off);
                int   oi = __shfl_xor(i2, off);
                if (ov > v || (ov == v && oi < i2)) { v = ov; i2 = oi; }
            }
            if (lane == 0) shLast = i2;
        }
        __syncthreads();
    }
}

// ---------------------------------------------------------------------------
// kernel 2: K=32 nearest neighbors per center — one wave per center
// ---------------------------------------------------------------------------
__global__ void k_knn(const float* __restrict__ pos, const int* __restrict__ wsIdx,
                      int* __restrict__ wsCol, float* __restrict__ wsBd) {
    const int wv = threadIdx.x >> 5, lane = threadIdx.x & 31;
    const int c  = blockIdx.x * 8 + wv;
    const int ci = wsIdx[c];
    const float cx = pos[ci*3+0], cy = pos[ci*3+1], cz = pos[ci*3+2];

    float bestd = 3.0e38f; int bestidx = -1;
    float curmax = 3.0e38f;

    for (int chunk = 0; chunk < N_PTS / 32; ++chunk) {
        int p = chunk * 32 + lane;
        float dx = pos[p*3+0]-cx, dy = pos[p*3+1]-cy, dz = pos[p*3+2]-cz;
        float d2 = dx*dx + dy*dy + dz*dz;
        unsigned long long b = __ballot(d2 < curmax);
        while (b) {
            int l = __ffsll(b) - 1; b &= b - 1;
            float cd = __shfl(d2, l);
            if (cd < curmax) {                       // uniform branch
                float v = bestd; int li = lane;      // evict current max slot
#pragma unroll
                for (int off = 16; off; off >>= 1) {
                    float ov = __shfl_xor(v, off);
                    int   ol = __shfl_xor(li, off);
                    if (ov > v || (ov == v && ol < li)) { v = ov; li = ol; }
                }
                if (lane == li) { bestd = cd; bestidx = chunk * 32 + l; }
                float mx = bestd;
#pragma unroll
                for (int off = 16; off; off >>= 1)
                    mx = fmaxf(mx, __shfl_xor(mx, off));
                curmax = mx;
            }
        }
    }
    wsCol[c * K_NBR + lane] = bestidx;
    wsBd [c * K_NBR + lane] = bestd;
}

// ---------------------------------------------------------------------------
// kernel 3: edge features (fragment-order LDS) -> WMMA GEMM1 -> relu/mask ->
//           segment max -> agg written in kernel-4 A-fragment order
// block = 256 threads (8 waves), 4 centers (128 edge rows)
// dynamic LDS: eaA[20480]f16 | w1s[20480]f16 | b1s[128]f32 | vmsk[128]i32 | part[8*128]f32
// ---------------------------------------------------------------------------
__global__ void k_mlp1(const float* __restrict__ x, const float* __restrict__ pos,
                       const float* __restrict__ nrm, const float* __restrict__ b1,
                       const int* __restrict__ wsIdx, const int* __restrict__ wsCol,
                       const float* __restrict__ wsBd, const _Float16* __restrict__ W1f,
                       _Float16* __restrict__ aggf) {
    extern __shared__ char sm3[];
    _Float16* ea   = (_Float16*)(sm3);            // 40960 B, A fragments
    _Float16* w1s  = (_Float16*)(sm3 + 40960);    // 40960 B, B fragments
    float*    b1s  = (float*)   (sm3 + 81920);
    int*      vmsk = (int*)     (sm3 + 82432);
    float*    part = (float*)   (sm3 + 82944);

    const int t = threadIdx.x;
    const int cBase = blockIdx.x * 4;

    // ---- async-stage W1 fragments into LDS (40960 B, 2560 x 16B chunks) ----
    {
        unsigned ldsBase = (unsigned)(uintptr_t)w1s;
        unsigned long long gbase = (unsigned long long)(uintptr_t)W1f;
#pragma unroll
        for (int it = 0; it < 10; ++it) {
            unsigned off = (unsigned)(t * 16 + it * 4096);
            ASYNC_LD_B128(ldsBase + off, off, gbase);
        }
    }
    for (int i = t; i < H_DIM; i += 256) b1s[i] = b1[i];

    // ---- build edge-feature A tile in fragment order ----
    {
        const int row  = t >> 1, prt = t & 1;
        const int ci   = row >> 5, slot = row & 31;
        const int cg   = cBase + ci;
        const int e1   = wsIdx[cg];
        const int e0   = wsCol[cg * K_NBR + slot];
        if (prt == 0) vmsk[row] = (wsBd[cg * K_NBR + slot] <= R2) ? 1 : 0;

        float4 ppf = make_float4(0.f, 0.f, 0.f, 0.f);
        if (prt == 1) {
            float px = pos[e0*3+0]-pos[e1*3+0];
            float py = pos[e0*3+1]-pos[e1*3+1];
            float pz = pos[e0*3+2]-pos[e1*3+2];
            float n0x = nrm[e0*3+0], n0y = nrm[e0*3+1], n0z = nrm[e0*3+2];
            float n1x = nrm[e1*3+0], n1y = nrm[e1*3+1], n1z = nrm[e1*3+2];
            ppf.x = safe_norm3(px, py, pz);
            ppf.y = get_angle3(n1x, n1y, n1z, px, py, pz);
            ppf.z = get_angle3(n0x, n0y, n0z, px, py, pz);
            ppf.w = get_angle3(n1x, n1y, n1z, n0x, n0y, n0z);
        }
        const int j0 = prt * 80;
#pragma unroll
        for (int jq = 0; jq < 20; ++jq) {
            int j = j0 + jq * 4;
            float4 f;
            if (j < D_FEAT)            f = *(const float4*)(x + e1 * D_FEAT + j);
            else if (j < 2 * D_FEAT)   f = *(const float4*)(x + e0 * D_FEAT + (j - D_FEAT));
            else if (j == 2 * D_FEAT)  f = ppf;
            else                       f = make_float4(0.f, 0.f, 0.f, 0.f);
            union { _Float16 h[4]; int2 v2; } u;
            u.h[0] = (_Float16)f.x; u.h[1] = (_Float16)f.y;
            u.h[2] = (_Float16)f.z; u.h[3] = (_Float16)f.w;
            *(int2*)(ea + eaA_addr(row, j, 5)) = u.v2;
        }
    }
    ASYNC_WAIT();
    __syncthreads();

    // ---- WMMA GEMM: wave w owns M-tile rows [16w, 16w+16) ----
    const int w = t >> 5, lane = t & 31;
    const int ln = lane & 15;
    const bool hi = lane >= 16;

    int vflag[8];
    {
        const int rbase = 16 * w + (hi ? 8 : 0);
#pragma unroll
        for (int r = 0; r < 8; ++r) vflag[r] = vmsk[rbase + r];
    }
    const int4* aq = (const int4*)ea  + ((w * 5) * 32 + lane) * 2;   // +64 per kc
    const int4* bq0 = (const int4*)w1s;

    for (int nt = 0; nt < 8; ++nt) {
        v8f acc = {};
        const int4* bq = bq0 + ((nt * 5) * 32 + lane) * 2;           // +64 per kc
#pragma unroll
        for (int kc = 0; kc < 5; ++kc) {
            union { v16h v; int4 q[2]; } A, B;
            A.q[0] = aq[kc * 64];  A.q[1] = aq[kc * 64 + 1];
            B.q[0] = bq[kc * 64];  B.q[1] = bq[kc * 64 + 1];
            acc = __builtin_amdgcn_wmma_f32_16x16x32_f16(
                      false, A.v, false, B.v, (short)0, acc, false, false);
        }
        const float bb = b1s[nt * 16 + ln];
        float mx = NEGV;
#pragma unroll
        for (int r = 0; r < 8; ++r) {
            float v = acc[r] + bb;
            v = v > 0.f ? v : 0.f;      // relu
            v = vflag[r] ? v : NEGV;    // radius mask
            mx = fmaxf(mx, v);
        }
        mx = fmaxf(mx, __shfl_xor(mx, 16));   // fold both lane halves (16 rows)
        if (!hi) part[w * H_DIM + nt * 16 + ln] = mx;
    }
    __syncthreads();

    // combine the two 16-row tiles of each center, empty -> 0,
    // emit f16 agg directly in kernel-4 A-fragment order
    for (int i = t; i < 4 * H_DIM; i += 256) {
        int c = i >> 7, n = i & 127;
        float v = fmaxf(part[(2 * c) * H_DIM + n], part[(2 * c + 1) * H_DIM + n]);
        v = (v > NEGV * 0.5f) ? v : 0.f;
        int row = cBase + c;
        int blk = row >> 7;
        aggf[blk * (128 * H_DIM) + eaA_addr(row & 127, n, 4)] = (_Float16)v;
    }
}

// ---------------------------------------------------------------------------
// kernel 4: out = agg @ W2 + b2 (WMMA f16, f32 out) — 128 centers per block
// dynamic LDS: aT[16384]f16 | w2s[16384]f16 | b2s[128]f32  (all fragment order)
// ---------------------------------------------------------------------------
__global__ void k_gemm2(const _Float16* __restrict__ aggf, const _Float16* __restrict__ W2f,
                        const float* __restrict__ b2, float* __restrict__ out) {
    extern __shared__ char sm4[];
    _Float16* aT  = (_Float16*)(sm4);             // 32768 B
    _Float16* w2s = (_Float16*)(sm4 + 32768);     // 32768 B
    float*    b2s = (float*)   (sm4 + 65536);

    const int t = threadIdx.x;
    const int rowsBase = blockIdx.x * 128;

    {   // async-stage this block's agg tile (32 KB) and W2 fragments (32 KB)
        unsigned aLds = (unsigned)(uintptr_t)aT;
        unsigned wLds = (unsigned)(uintptr_t)w2s;
        unsigned long long aG = (unsigned long long)(uintptr_t)(aggf + (size_t)blockIdx.x * 128 * H_DIM);
        unsigned long long wG = (unsigned long long)(uintptr_t)W2f;
#pragma unroll
        for (int it = 0; it < 8; ++it) {
            unsigned off = (unsigned)(t * 16 + it * 4096);
            ASYNC_LD_B128(aLds + off, off, aG);
            ASYNC_LD_B128(wLds + off, off, wG);
        }
    }
    for (int i = t; i < OUT_DIM; i += 256) b2s[i] = b2[i];
    ASYNC_WAIT();
    __syncthreads();

    const int w = t >> 5, lane = t & 31;
    const int ln = lane & 15;
    const bool hi = lane >= 16;
    const int4* aq  = (const int4*)aT  + ((w * 4) * 32 + lane) * 2;  // +64 per kc
    const int4* bq0 = (const int4*)w2s;

    for (int nt = 0; nt < 8; ++nt) {
        v8f acc = {};
        const int4* bq = bq0 + ((nt * 4) * 32 + lane) * 2;           // +64 per kc
#pragma unroll
        for (int kc = 0; kc < 4; ++kc) {
            union { v16h v; int4 q[2]; } A, B;
            A.q[0] = aq[kc * 64];  A.q[1] = aq[kc * 64 + 1];
            B.q[0] = bq[kc * 64];  B.q[1] = bq[kc * 64 + 1];
            acc = __builtin_amdgcn_wmma_f32_16x16x32_f16(
                      false, A.v, false, B.v, (short)0, acc, false, false);
        }
        const float bb = b2s[nt * 16 + ln];
#pragma unroll
        for (int r = 0; r < 8; ++r) {
            int rowAbs = rowsBase + 16 * w + (hi ? 8 : 0) + r;
            out[(size_t)rowAbs * OUT_DIM + nt * 16 + ln] = acc[r] + bb;
        }
    }
}

// ---------------------------------------------------------------------------
// kernel 5: tail — centers (pos[idx]) and idx (as float) into concat output
// ---------------------------------------------------------------------------
__global__ void k_tail(const float* __restrict__ pos, const int* __restrict__ wsIdx,
                       float* __restrict__ out) {
    int m = blockIdx.x * blockDim.x + threadIdx.x;
    if (m < M_CTR) {
        int id = wsIdx[m];
        float* cbase = out + (size_t)M_CTR * OUT_DIM;
        cbase[m * 3 + 0] = pos[id * 3 + 0];
        cbase[m * 3 + 1] = pos[id * 3 + 1];
        cbase[m * 3 + 2] = pos[id * 3 + 2];
        out[(size_t)M_CTR * OUT_DIM + (size_t)M_CTR * 3 + m] = (float)id;
    }
}

// ---------------------------------------------------------------------------
extern "C" void kernel_launch(void* const* d_in, const int* in_sizes, int n_in,
                              void* d_out, int out_size, void* d_ws, size_t ws_size,
                              hipStream_t stream) {
    const float* x   = (const float*)d_in[0];
    const float* pos = (const float*)d_in[1];
    const float* nrm = (const float*)d_in[2];
    // d_in[3] = batch (all zeros, unused)
    const float* W1  = (const float*)d_in[4];
    const float* b1  = (const float*)d_in[5];
    const float* W2  = (const float*)d_in[6];
    const float* b2  = (const float*)d_in[7];
    float* out = (float*)d_out;

    char* ws = (char*)d_ws;
    int*       wsIdx = (int*)      (ws + OFF_IDX);
    int*       wsCol = (int*)      (ws + OFF_COL);
    float*     wsBd  = (float*)    (ws + OFF_BD);
    _Float16*  W1f   = (_Float16*) (ws + OFF_W1F);
    _Float16*  W2f   = (_Float16*) (ws + OFF_W2F);
    _Float16*  aggf  = (_Float16*) (ws + OFF_AGG);

    k_convert<<<(H_DIM * KPAD + OUT_DIM * H_DIM + 255) / 256, 256, 0, stream>>>(W1, W2, W1f, W2f);
    k_fps<<<1, 1024, 0, stream>>>(pos, wsIdx);
    k_knn<<<M_CTR / 8, 256, 0, stream>>>(pos, wsIdx, wsCol, wsBd);
    k_mlp1<<<M_CTR / 4, 256, 87040, stream>>>(x, pos, nrm, b1, wsIdx, wsCol, wsBd, W1f, aggf);
    k_gemm2<<<M_CTR / 128, 256, 66048, stream>>>(aggf, W2f, b2, out);
    k_tail<<<(M_CTR + 255) / 256, 256, 0, stream>>>(pos, wsIdx, out);
}
